// CustomSimpleGRU_54915451847141
// MI455X (gfx1250) — compile-verified, matrix operand order checked by reading, not run on hardware
//
#include <hip/hip_runtime.h>
#include <hip/hip_bf16.h>

#define BATCH 128
#define SEQ   512
#define INSZ  128
#define HID   1024

typedef __attribute__((ext_vector_type(16))) __bf16 v16bf;
typedef __attribute__((ext_vector_type(8)))  __bf16 v8bf;
typedef __attribute__((ext_vector_type(8)))  float  v8f;

#define KC 128                 // K-chunk staged in LDS per iteration
#define LDST (KC + 8)          // padded LDS row stride (bf16 elems) to avoid bank conflicts

static __device__ inline v16bf join8(v8bf lo, v8bf hi) {
    return __builtin_shufflevector(lo, hi, 0,1,2,3,4,5,6,7,8,9,10,11,12,13,14,15);
}

// ---------------------------------------------------------------------------
// Pack W (K x 3072, f32 row-major) -> Wzn (2048 x K, bf16), keeping only the
// z chunk (cols 0..1023) and n chunk (cols 2048..3071), TRANSPOSED so each
// WMMA B lane reads contiguous K. Row j<1024 -> z col j ; row j>=1024 -> n col j+1024.
// ---------------------------------------------------------------------------
__global__ __launch_bounds__(256) void pack_w_kernel(const float* __restrict__ W,
                                                     __bf16* __restrict__ out, int K) {
    long idx = (long)blockIdx.x * 256 + threadIdx.x;
    long total = 2048L * K;
    if (idx >= total) return;
    int j = (int)(idx / K);
    int k = (int)(idx % K);
    int col = (j < 1024) ? j : (j + 1024);
    out[idx] = (__bf16)W[(long)k * 3072 + col];
}

__global__ __launch_bounds__(256) void cvt_bf16_kernel(const float* __restrict__ in,
                                                       __bf16* __restrict__ out, long n) {
    long i = (long)blockIdx.x * 256 + threadIdx.x;
    if (i < n) out[i] = (__bf16)in[i];
}

// ---------------------------------------------------------------------------
// One GRU step for one layer, fused GEMM(z,n) + activation + state update.
// A = [A0 (K0 cols) | h_prev_bf (1024 cols)], 128 rows. Grid: 64 WGs x 256 thr.
// Each WG: 16 rows x 128 cols (8 waves, one 16-col tile each). Each wave
// accumulates its z tile and its n tile with v_wmma_f32_16x16x32_bf16.
// ---------------------------------------------------------------------------
__global__ __launch_bounds__(256)
void gru_step_kernel(const __bf16* __restrict__ A0, int lda0, int K0,
                     const __bf16* __restrict__ hprev_bf,  // [128][1024]
                     const __bf16* __restrict__ Wzn,       // [2048][K] transposed bf16
                     const float*  __restrict__ bias,      // [3*1024]: z at j, n at 2048+j
                     float*        __restrict__ hstate,    // [128][1024] f32 in/out
                     __bf16*       __restrict__ hout_bf,   // [128][1024] next-step operand
                     int K)                                // K0 + 1024
{
    __shared__ __bf16 As[16 * LDST];

    const int tid      = threadIdx.x;
    const int wave     = tid >> 5;
    const int lane     = tid & 31;
    const int row_tile = blockIdx.x >> 3;          // 0..7
    const int colgrp   = blockIdx.x & 7;           // 0..7
    const int j0       = (colgrp * 8 + wave) * 16; // this wave's column-tile base
    const int row0     = row_tile * 16;

    const int ncol  = lane & 15;       // WMMA B/C column within tile
    const int khalf = (lane >> 4) * 8; // K sub-offset per B/A layout

    const __bf16* Bz = Wzn + (long)(j0 + ncol) * K;          // z-chunk row (transposed)
    const __bf16* Bn = Wzn + (long)(1024 + j0 + ncol) * K;   // n-chunk row

    v8f accz = {}; v8f accn = {};

    const int nchunks = K / KC;
    for (int c = 0; c < nchunks; ++c) {
        const int kbase = c * KC;

        // --- stage 16 x 128 bf16 slice of A into LDS (256 thr x 16B each) ---
        {
            const int r  = tid >> 4;          // 0..15 row
            const int kk = (tid & 15) * 8;    // 0..120
            const __bf16* src = (kbase < K0)
                ? (A0       + (long)(row0 + r) * lda0 + (kbase + kk))
                : (hprev_bf + (long)(row0 + r) * HID  + (kbase - K0 + kk));
            *(v8bf*)(&As[r * LDST + kk]) = *(const v8bf*)src;
        }
        __syncthreads();

        // prefetch next K-chunk of weights (L2-resident; emits global_prefetch)
        if (c + 1 < nchunks) {
            __builtin_prefetch(Bz + kbase + KC, 0, 1);
            __builtin_prefetch(Bn + kbase + KC, 0, 1);
        }

        // --- 4 WMMA k-steps of 32 over this chunk ---
        #pragma unroll
        for (int ks = 0; ks < KC; ks += 32) {
            const int m = lane & 15;
            v8bf alo = *(const v8bf*)(&As[m * LDST + ks + khalf]);
            v8bf ahi = *(const v8bf*)(&As[m * LDST + ks + khalf + 16]);
            v16bf a  = join8(alo, ahi);

            const int kk = kbase + ks + khalf;
            v16bf bz = join8(*(const v8bf*)(Bz + kk), *(const v8bf*)(Bz + kk + 16));
            v16bf bn = join8(*(const v8bf*)(Bn + kk), *(const v8bf*)(Bn + kk + 16));

            accz = __builtin_amdgcn_wmma_f32_16x16x32_bf16(false, a, false, bz,
                                                           (short)0, accz, false, false);
            accn = __builtin_amdgcn_wmma_f32_16x16x32_bf16(false, a, false, bn,
                                                           (short)0, accn, false, false);
        }
        __syncthreads();
    }

    // --- fused epilogue: z=sigmoid, n=tanh, h = (1-z)*n + z*h ---
    const int col = j0 + ncol;
    const float bz = bias[col];
    const float bn = bias[2048 + col];
    #pragma unroll
    for (int v = 0; v < 8; ++v) {
        const int row = row0 + (lane >> 4) * 8 + v;   // C-layout: M = (L/16)*8 + vgpr
        float zg = accz[v] + bz;
        zg = 1.0f / (1.0f + __expf(-zg));
        float ng = tanhf(accn[v] + bn);
        const long o = (long)row * HID + col;
        float hold = hstate[o];
        float hnew = (1.0f - zg) * ng + zg * hold;
        hstate[o]  = hnew;
        hout_bf[o] = (__bf16)hnew;
    }
}

// ---------------------------------------------------------------------------
// Final FC: out[b] = dot(h1[b], Wfc) + bfc  (128 rows x 1024)
// ---------------------------------------------------------------------------
__global__ __launch_bounds__(256)
void fc_kernel(const float* __restrict__ h, const float* __restrict__ Wfc,
               const float* __restrict__ bfc, float* __restrict__ out) {
    __shared__ float red[8];
    const int b = blockIdx.x;
    float s = 0.0f;
    for (int j = threadIdx.x; j < HID; j += 256)
        s += h[(long)b * HID + j] * Wfc[j];
    #pragma unroll
    for (int off = 16; off > 0; off >>= 1)
        s += __shfl_down(s, off, 32);
    if ((threadIdx.x & 31) == 0) red[threadIdx.x >> 5] = s;
    __syncthreads();
    if (threadIdx.x == 0) {
        float t = 0.0f;
        #pragma unroll
        for (int w = 0; w < 8; ++w) t += red[w];
        out[b] = t + bfc[0];
    }
}

// ---------------------------------------------------------------------------
extern "C" void kernel_launch(void* const* d_in, const int* in_sizes, int n_in,
                              void* d_out, int out_size, void* d_ws, size_t ws_size,
                              hipStream_t stream) {
    const float* x   = (const float*)d_in[0];
    const float* W0  = (const float*)d_in[1];
    const float* b0  = (const float*)d_in[2];
    const float* W1  = (const float*)d_in[3];
    const float* b1  = (const float*)d_in[4];
    const float* Wfc = (const float*)d_in[5];
    const float* bfc = (const float*)d_in[6];
    float* out = (float*)d_out;

    // workspace carving (256B aligned)
    char* ws = (char*)d_ws;
    auto carve = [&](size_t bytes) {
        char* p = ws;
        ws += (bytes + 255) & ~(size_t)255;
        return p;
    };
    const long XN   = (long)BATCH * SEQ * INSZ;
    const long HSZ  = (long)BATCH * HID;           // elems per h buffer
    __bf16* x_bf  = (__bf16*)carve(XN * 2);
    __bf16* Wzn0  = (__bf16*)carve(2048L * 1152 * 2);
    __bf16* Wzn1  = (__bf16*)carve(2048L * 2048 * 2);
    float*  h0f   = (float*) carve(HSZ * 4);
    float*  h1f   = (float*) carve(HSZ * 4);
    __bf16* h0bf  = (__bf16*)carve(2 * HSZ * 2);   // double-buffered
    __bf16* h1bf  = (__bf16*)carve(2 * HSZ * 2);

    // one-time prep (re-done every call: deterministic, inputs never mutated)
    pack_w_kernel<<<(int)((2048L*1152 + 255) / 256), 256, 0, stream>>>(W0, Wzn0, 1152);
    pack_w_kernel<<<(int)((2048L*2048 + 255) / 256), 256, 0, stream>>>(W1, Wzn1, 2048);
    cvt_bf16_kernel<<<(int)((XN + 255) / 256), 256, 0, stream>>>(x, x_bf, XN);
    hipMemsetAsync(h0f,  0, HSZ * 4, stream);
    hipMemsetAsync(h1f,  0, HSZ * 4, stream);
    hipMemsetAsync(h0bf, 0, 2 * HSZ * 2, stream);
    hipMemsetAsync(h1bf, 0, 2 * HSZ * 2, stream);

    for (int t = 0; t < SEQ; ++t) {
        const int rd = t & 1, wr = rd ^ 1;
        // layer 0: A = [x_t (128) | h0_prev (1024)], K = 1152
        gru_step_kernel<<<64, 256, 0, stream>>>(
            x_bf + (long)t * INSZ, SEQ * INSZ, INSZ,
            h0bf + (long)rd * HSZ, Wzn0, b0, h0f, h0bf + (long)wr * HSZ, 1152);
        // layer 1: A = [h0_new (1024) | h1_prev (1024)], K = 2048
        gru_step_kernel<<<64, 256, 0, stream>>>(
            h0bf + (long)wr * HSZ, HID, HID,
            h1bf + (long)rd * HSZ, Wzn1, b1, h1f, h1bf + (long)wr * HSZ, 2048);
    }

    fc_kernel<<<BATCH, 256, 0, stream>>>(h1f, Wfc, bfc, out);
}
